// model5_27814208209096
// MI455X (gfx1250) — matrix operand order repaired and checked by reading, not compiled, not run on hardware
//
#include <hip/hip_runtime.h>
#include <hip/hip_bf16.h>
#include <cstdint>

// ---------------------------------------------------------------------------
// Factorial HMM (pyro model_5) forward log-likelihood on MI455X (gfx1250).
//   1) emission GEMM emit[BT x 1024] = W4[BT x 384] * T4[384 x 1024], f16 WMMA
//      with both operands pre-packed in global memory in exact WMMA fragment
//      order -> per-lane 32B contiguous loads, no LDS staging.
//   2) forward recursion: alpha' = m + log(Pw^T * exp(alpha-m) * Px) + emit_t
//      via V_WMMA_F32_16X16X4_F32; Px tile preloaded into LDS with the
//      Tensor Data Mover (tensor_load_to_lds + s_wait_tensorcnt).
// ---------------------------------------------------------------------------

#define NEG_INF (-1e30f)
#define BATCH 64
#define TSEQ  512
#define DDIM  88
#define HDIM  32
#define NSTATE 1024   // H*H
#define KDIM  384     // 4*88 padded to multiple of 32
#define NKB   (KDIM / 32)
#define TC    128     // time chunk
#define NCHUNK (TSEQ / TC)

typedef float    v2f  __attribute__((ext_vector_type(2)));
typedef float    v8f  __attribute__((ext_vector_type(8)));
typedef _Float16 v16h __attribute__((ext_vector_type(16)));
typedef uint32_t v4u  __attribute__((ext_vector_type(4)));
typedef uint32_t v8u  __attribute__((ext_vector_type(8)));

// A-fragment K index for (vgpr v, lane-half h)  [ISA 7.12.2, 16-bit A 16x32]
__device__ __forceinline__ int slotOf(int koE) {
    // inverse of ka(v,h) = (v<4 ? 2v : 16+2(v-4)) + 8h for even koE in [0,32)
    int hh = (koE >> 3) & 1;
    int vv = ((koE & 7) >> 1) + ((koE >> 4) << 2);
    return hh * 16 * 8 + vv;     // lane-half offset*8 + v, caller adds rr*8
}

// ---------------------------------------------------------------------------
// Pack A (one-hot W4) directly into WMMA A-fragment order:
//   AF[((btTile*12 + kb32)*32 + lane)*8 + v] = uint holding halves (K, K+1),
//   K = kb32*32 + ka(v,h), row = btTile*16 + (lane&15).
// ---------------------------------------------------------------------------
__global__ __launch_bounds__(256) void pack_A(const float* __restrict__ seq,
                                              const int* __restrict__ lengths,
                                              uint32_t* __restrict__ AF) {
    int idx = blockIdx.x * 256 + threadIdx.x;       // 32768 * 96 threads
    int bt = idx / 96;
    int d  = idx - bt * 96;                          // 0..95 (>=88: K padding)
    int b  = bt >> 9;
    int t  = bt & 511;
    uint32_t lo = 0u, hi = 0u;                       // pairs (k=0,1) and (k=2,3)
    if (d < DDIM && t < lengths[b]) {
        float y  = seq[((size_t)b * TSEQ + t) * DDIM + d];
        float yp = (t > 0) ? seq[((size_t)b * TSEQ + t - 1) * DDIM + d] : 0.0f;
        int k = ((yp > 0.5f) ? 2 : 0) + ((y > 0.5f) ? 1 : 0);
        const uint32_t one = 0x3C00u;                // f16 1.0
        if      (k == 0) lo = one;
        else if (k == 1) lo = one << 16;
        else if (k == 2) hi = one;
        else             hi = one << 16;
    }
    int btTile = bt >> 4, rr = bt & 15;
    int kd0 = d * 4;
    int kb32 = kd0 >> 5, ko = kd0 & 31;
    size_t base = (((size_t)btTile * NKB + kb32) * 32) * 8 + (size_t)rr * 8;
    AF[base + slotOf(ko)]     = lo;
    AF[base + slotOf(ko + 2)] = hi;
}

// ---------------------------------------------------------------------------
// Pack B (log-prob table T4) into WMMA B-fragment order:
//   BF[((kb32*64 + n16)*32 + lane)*8 + v] = halves (K,K+1) for
//   K = kb32*32 + ka(v,h), col = n16*16 + (lane&15).
// ---------------------------------------------------------------------------
__global__ __launch_bounds__(256) void pack_B(const float* __restrict__ py,
                                              uint32_t* __restrict__ BF) {
    int idx  = blockIdx.x * 256 + threadIdx.x;       // 12*64*32*8 = 196608
    int v    = idx & 7;
    int lane = (idx >> 3) & 31;
    int n16  = (idx >> 8) & 63;
    int kb32 = idx >> 14;
    int h = lane >> 4, rr = lane & 15;
    int col = n16 * 16 + rr;
    uint32_t out = 0u;
#pragma unroll
    for (int e = 0; e < 2; ++e) {
        int kd = kb32 * 32 + ((v < 4) ? 2 * v : 16 + 2 * (v - 4)) + 8 * h + e;
        float val = 0.0f;
        if (kd < 4 * DDIM) {
            int d = kd >> 2, k = kd & 3, j = k >> 1;
            float p = py[((size_t)col * 2 + j) * DDIM + d];
            val = (k & 1) ? __logf(p) : __logf(1.0f - p);
        }
        _Float16 hf = (_Float16)val;
        uint16_t bits;
        __builtin_memcpy(&bits, &hf, 2);
        out |= (uint32_t)bits << (16 * e);
    }
    BF[idx] = out;
}

// ---------------------------------------------------------------------------
// Emission GEMM, one time chunk. 256 threads = 8 waves (2x4 grid), each wave
// owns a 32x32 output tile (2x2 WMMA tiles). Fragments load straight from
// global in fragment order: 2x global_load_b128 per fragment.
// ---------------------------------------------------------------------------
__global__ __launch_bounds__(256) void emit_gemm(const uint32_t* __restrict__ AF,
                                                 const uint32_t* __restrict__ BF,
                                                 float* __restrict__ emitC,
                                                 int t0) {
    const int tid  = threadIdx.x;
    const int wv   = tid >> 5, lane = tid & 31;
    const int h    = lane >> 4, rr = lane & 15;
    const int mw   = wv >> 2,  nw = wv & 3;          // 2x4 wave grid
    const int nBlock = blockIdx.x * 128;
    const int mBlock = blockIdx.y * 64;

    // Global bt-tile index of this wave's two A row-tiles (TC multiple of 16).
    int btTile[2];
#pragma unroll
    for (int i = 0; i < 2; ++i) {
        int m0 = mBlock + mw * 32 + i * 16;          // chunk-local row base
        int b  = m0 >> 7;                            // / TC
        int tl = m0 & (TC - 1);
        btTile[i] = b * (TSEQ / 16) + ((t0 + tl) >> 4);
    }
    const int n16a = (nBlock >> 4) + nw * 2;

    const v8f zero = {0, 0, 0, 0, 0, 0, 0, 0};
    v8f acc00 = zero, acc01 = zero, acc10 = zero, acc11 = zero;

    for (int kb32 = 0; kb32 < NKB; ++kb32) {
        union FU { uint4 q[2]; v16h v; } af0, af1, bf0, bf1;
        {
            const uint32_t* p0 = AF + (((size_t)btTile[0] * NKB + kb32) * 32 + lane) * 8;
            const uint32_t* p1 = AF + (((size_t)btTile[1] * NKB + kb32) * 32 + lane) * 8;
            af0.q[0] = *(const uint4*)p0; af0.q[1] = *(const uint4*)(p0 + 4);
            af1.q[0] = *(const uint4*)p1; af1.q[1] = *(const uint4*)(p1 + 4);
        }
        {
            const uint32_t* p0 = BF + (((size_t)kb32 * 64 + n16a) * 32 + lane) * 8;
            const uint32_t* p1 = BF + (((size_t)kb32 * 64 + n16a + 1) * 32 + lane) * 8;
            bf0.q[0] = *(const uint4*)p0; bf0.q[1] = *(const uint4*)(p0 + 4);
            bf1.q[0] = *(const uint4*)p1; bf1.q[1] = *(const uint4*)(p1 + 4);
        }
        acc00 = __builtin_amdgcn_wmma_f32_16x16x32_f16(false, af0.v, false, bf0.v, (short)0, acc00, false, false);
        acc01 = __builtin_amdgcn_wmma_f32_16x16x32_f16(false, af0.v, false, bf1.v, (short)0, acc01, false, false);
        acc10 = __builtin_amdgcn_wmma_f32_16x16x32_f16(false, af1.v, false, bf0.v, (short)0, acc10, false, false);
        acc11 = __builtin_amdgcn_wmma_f32_16x16x32_f16(false, af1.v, false, bf1.v, (short)0, acc11, false, false);
    }

    // C/D layout: VGPR v, lanes 0-15 -> M=v, lanes 16-31 -> M=v+8, N=lane%16
#pragma unroll
    for (int v = 0; v < 8; ++v) {
        int r0 = mBlock + mw * 32 + v + 8 * h;
        int cA = nBlock + nw * 32 + rr;
        emitC[(size_t)r0 * NSTATE + cA]             = acc00[v];
        emitC[(size_t)r0 * NSTATE + cA + 16]        = acc01[v];
        emitC[(size_t)(r0 + 16) * NSTATE + cA]      = acc10[v];
        emitC[(size_t)(r0 + 16) * NSTATE + cA + 16] = acc11[v];
    }
}

// ---------------------------------------------------------------------------
// Forward recursion, one chunk of TC steps. One block (4 waves) per batch.
// alpha' = m + log(PwT * exp(alpha - m) * Px) + emit_t, frozen past length.
// Px preloaded into LDS by the Tensor Data Mover.
// ---------------------------------------------------------------------------
__global__ __launch_bounds__(128) void fwd_chunk(const float* __restrict__ emitC,
                                                 const int* __restrict__ lengths,
                                                 const float* __restrict__ pw,
                                                 const float* __restrict__ px,
                                                 float* __restrict__ alphaBuf,
                                                 float* __restrict__ out,
                                                 const float* __restrict__ scale,
                                                 int ch) {
    __shared__ float PxS[NSTATE], PwTS[NSTATE];
    __shared__ float alpha_s[NSTATE], Es[NSTATE], tmp_s[NSTATE];
    __shared__ float red_s[128];

    const int tid = threadIdx.x;
    const int b   = blockIdx.x;
    const int t0  = ch * TC;

    // --- TDM: DMA the 32x32 f32 Px tile into LDS (wave 0 issues) ---
    if (tid < 32) {
        uint64_t gpx = (uint64_t)(uintptr_t)px;
        uint32_t ldsOff =
            (uint32_t)(size_t)(__attribute__((address_space(3))) float*)PxS;
        v4u g0;
        g0[0] = 1u;                                   // count=1 (user mode)
        g0[1] = ldsOff;                               // lds_addr
        g0[2] = (uint32_t)gpx;                        // global_addr[31:0]
        g0[3] = (uint32_t)((gpx >> 32) & 0x01FFFFFFu) // global_addr[56:32]
              | (2u << 30);                           // type=2 ("image")
        v8u g1;
        g1[0] = (2u << 16);                           // data_size = 4 bytes
        g1[1] = (32u << 16);                          // tensor_dim0 = 32
        g1[2] = (32u << 16);                          // tensor_dim1 = 32
        g1[3] = (32u << 16);                          // tile_dim0   = 32
        g1[4] = 32u;                                  // tile_dim1   = 32
        g1[5] = 32u;                                  // tensor_dim0_stride = 32
        g1[6] = 0u;
        g1[7] = 0u;
        asm volatile("tensor_load_to_lds %0, %1" :: "s"(g0), "s"(g1) : "memory");
        __builtin_amdgcn_s_wait_tensorcnt(0);
    }
    for (int i = tid; i < NSTATE; i += 128) {
        int w = i >> 5, wp = i & 31;
        PwTS[i] = pw[wp * HDIM + w];                  // PwT[w][w'] = Pw[w'][w]
        alpha_s[i] = (ch == 0) ? ((i == 0) ? 0.0f : NEG_INF)
                               : alphaBuf[(size_t)b * NSTATE + i];
    }
    __syncthreads();

    const int len = lengths[b];
    const float* eBase = emitC + (size_t)b * TC * NSTATE;

    const int wv = tid >> 5, lane = tid & 31;
    const int h  = lane >> 4, rr = lane & 15;
    const int it = wv >> 1, jt = wv & 1;              // 2x2 tile grid over 32x32

    for (int tl = 0; tl < TC; ++tl) {
        if (t0 + tl >= len) break;                    // uniform per block

        // --- m = max(alpha) ---
        float pm = -3.0e38f;
        for (int i = tid; i < NSTATE; i += 128) pm = fmaxf(pm, alpha_s[i]);
        red_s[tid] = pm; __syncthreads();
        for (int s = 64; s > 0; s >>= 1) {
            if (tid < s) red_s[tid] = fmaxf(red_s[tid], red_s[tid + s]);
            __syncthreads();
        }
        const float m = red_s[0];

        // --- E = exp(alpha - m) ---
        for (int i = tid; i < NSTATE; i += 128) Es[i] = __expf(alpha_s[i] - m);
        __syncthreads();

        // --- tmp = E x Px (eliminate x') ---
        {
            v8f acc = {0, 0, 0, 0, 0, 0, 0, 0};
#pragma unroll
            for (int ko = 0; ko < HDIM; ko += 4) {
                v2f a, bf;
                a.x  = Es[(it * 16 + rr) * HDIM + ko + 2 * h];
                a.y  = Es[(it * 16 + rr) * HDIM + ko + 2 * h + 1];
                bf.x = PxS[(ko + 2 * h) * HDIM + jt * 16 + rr];
                bf.y = PxS[(ko + 2 * h + 1) * HDIM + jt * 16 + rr];
                acc = __builtin_amdgcn_wmma_f32_16x16x4_f32(false, a, false, bf, (short)0, acc, false, false);
            }
#pragma unroll
            for (int v = 0; v < 8; ++v)
                tmp_s[(it * 16 + v + 8 * h) * HDIM + jt * 16 + rr] = acc[v];
        }
        __syncthreads();

        // --- G = PwT x tmp (eliminate w'), fused log + emit epilogue ---
        {
            v8f acc = {0, 0, 0, 0, 0, 0, 0, 0};
#pragma unroll
            for (int ko = 0; ko < HDIM; ko += 4) {
                v2f a, bf;
                a.x  = PwTS[(it * 16 + rr) * HDIM + ko + 2 * h];
                a.y  = PwTS[(it * 16 + rr) * HDIM + ko + 2 * h + 1];
                bf.x = tmp_s[(ko + 2 * h) * HDIM + jt * 16 + rr];
                bf.y = tmp_s[(ko + 2 * h + 1) * HDIM + jt * 16 + rr];
                acc = __builtin_amdgcn_wmma_f32_16x16x4_f32(false, a, false, bf, (short)0, acc, false, false);
            }
            const float* eRow = eBase + (size_t)tl * NSTATE;
#pragma unroll
            for (int v = 0; v < 8; ++v) {
                int idx = (it * 16 + v + 8 * h) * HDIM + jt * 16 + rr;
                alpha_s[idx] = m + __logf(acc[v]) + eRow[idx];
            }
        }
        __syncthreads();
    }

    // persist alpha for next chunk
    for (int i = tid; i < NSTATE; i += 128)
        alphaBuf[(size_t)b * NSTATE + i] = alpha_s[i];

    if (ch == NCHUNK - 1) {
        __syncthreads();
        float pm = -3.0e38f;
        for (int i = tid; i < NSTATE; i += 128) pm = fmaxf(pm, alpha_s[i]);
        red_s[tid] = pm; __syncthreads();
        for (int s = 64; s > 0; s >>= 1) {
            if (tid < s) red_s[tid] = fmaxf(red_s[tid], red_s[tid + s]);
            __syncthreads();
        }
        const float m2 = red_s[0];
        __syncthreads();
        float ps = 0.0f;
        for (int i = tid; i < NSTATE; i += 128) ps += __expf(alpha_s[i] - m2);
        red_s[tid] = ps; __syncthreads();
        for (int s = 64; s > 0; s >>= 1) {
            if (tid < s) red_s[tid] += red_s[tid + s];
            __syncthreads();
        }
        if (tid == 0) atomicAdd(out, scale[0] * (m2 + __logf(red_s[0])));
    }
}

// ---------------------------------------------------------------------------
// Workspace layout (bytes):
//   AF  u32-packed f16 frags [2048*12*32*8]  25,165,824
//   BF  u32-packed f16 frags [12*64*32*8]       786,432
//   emit f32 [64*TC x 1024]                  33,554,432
//   alpha f32 [64 x 1024]                       262,144    total ~59.8 MB
// ---------------------------------------------------------------------------
extern "C" void kernel_launch(void* const* d_in, const int* in_sizes, int n_in,
                              void* d_out, int out_size, void* d_ws, size_t ws_size,
                              hipStream_t stream) {
    const float* seq = (const float*)d_in[0];
    const int*   len = (const int*)d_in[1];
    const float* pw  = (const float*)d_in[2];
    const float* px  = (const float*)d_in[3];
    const float* py  = (const float*)d_in[4];
    const float* scl = (const float*)d_in[5];
    float* out = (float*)d_out;

    char* ws = (char*)d_ws;
    const size_t offA  = 0;
    const size_t offB  = offA + (size_t)BATCH * TSEQ * KDIM * 2;
    const size_t offE  = offB + (size_t)KDIM * NSTATE * 2;
    const size_t offAl = offE + (size_t)BATCH * TC * NSTATE * 4;
    uint32_t* AF    = (uint32_t*)(ws + offA);
    uint32_t* BF    = (uint32_t*)(ws + offB);
    float*    emitC = (float*)(ws + offE);
    float*    alphaB = (float*)(ws + offAl);

    hipMemsetAsync(out, 0, sizeof(float), stream);

    pack_A<<<(BATCH * TSEQ * 96) / 256, 256, 0, stream>>>(seq, len, AF);
    pack_B<<<(NKB * 64 * 32 * 8) / 256, 256, 0, stream>>>(py, BF);

    for (int ch = 0; ch < NCHUNK; ++ch) {
        emit_gemm<<<dim3(NSTATE / 128, (BATCH * TC) / 64), 256, 0, stream>>>(
            AF, BF, emitC, ch * TC);
        fwd_chunk<<<BATCH, 128, 0, stream>>>(emitC, len, pw, px, alphaB, out, scl, ch);
    }
}
